// ClusteredLinformerAttention_17867063951750
// MI455X (gfx1250) — compile-verified, matrix operand order checked by reading, not run on hardware
//
#include <hip/hip_runtime.h>
#include <math.h>

// Problem constants (from reference)
#define B_      16
#define S_      2048
#define DM_     512
#define H_      8
#define P_      128
#define C_      16
#define DEPTH_  64

typedef __attribute__((ext_vector_type(16))) __bf16       bf16x16;
typedef __attribute__((ext_vector_type(8)))  float        floatx8;
typedef __attribute__((ext_vector_type(8)))  unsigned int uintx8;
typedef __attribute__((ext_vector_type(4)))  unsigned int uint4v;
typedef __attribute__((ext_vector_type(8)))  int          int8v;
typedef __attribute__((ext_vector_type(4)))  int          int4v;

__device__ __forceinline__ unsigned short f2bf(float f) {
  unsigned int u = __builtin_bit_cast(unsigned int, f);
  u += 0x7FFFu + ((u >> 16) & 1u);          // round-to-nearest-even
  return (unsigned short)(u >> 16);
}
__device__ __forceinline__ float bf2f(unsigned short h) {
  unsigned int u = ((unsigned int)h) << 16;
  return __builtin_bit_cast(float, u);
}

// ---------------- TDM: 2-D bf16 tile load (global -> LDS), D# per ISA 08 ----
// tile: tileX contiguous elems per row, tileY rows, row stride strideX elems.
// Tensor described as exactly the tile (never OOB for our shapes).
__device__ __forceinline__ void tdm_load_2d(const void* gptr, unsigned lds_off,
                                            unsigned tileX, unsigned tileY,
                                            unsigned long long strideX) {
  unsigned long long ga = (unsigned long long)gptr;
  uint4v g0;
  g0.x = 1u;                                               // count=1, user D#
  g0.y = lds_off;                                          // lds_addr (bytes)
  g0.z = (unsigned)(ga & 0xFFFFFFFFu);                     // global_addr[31:0]
  g0.w = (unsigned)((ga >> 32) & 0x1FFFFFFu) | (2u << 30); // [56:32] | type=2
  int8v g1;
  g1[0] = (int)(1u << 16);                                 // data_size = 2 B
  g1[1] = (int)((tileX & 0xFFFFu) << 16);                  // tensor_dim0[15:0]
  g1[2] = (int)(((tileX >> 16) & 0xFFFFu) | ((tileY & 0xFFFFu) << 16)); // td0hi|td1lo
  g1[3] = (int)(((tileY >> 16) & 0xFFFFu) | ((tileX & 0xFFFFu) << 16)); // td1hi|tile_dim0
  g1[4] = (int)(tileY & 0xFFFFu);                          // tile_dim1, tile_dim2=0
  g1[5] = (int)(strideX & 0xFFFFFFFFull);                  // dim0_stride[31:0]
  g1[6] = (int)((strideX >> 32) & 0xFFFFull);              // dim0_stride[47:32]
  g1[7] = 0;
  int4v gz4 = {0, 0, 0, 0};
  int8v gz8 = {0, 0, 0, 0, 0, 0, 0, 0};
  // 6-arg form (clang-23 / therock-10.0 toolchain)
  __builtin_amdgcn_tensor_load_to_lds(g0, g1, gz4, gz4, gz8, 0);
}

// ---------------------------------------------------------------- converters
__global__ void cvt_f32_bf16(const float* __restrict__ in,
                             unsigned short* __restrict__ out, int n) {
  int i = blockIdx.x * blockDim.x + threadIdx.x;
  if (i < n) out[i] = f2bf(in[i]);
}

// ------------------------------------------------------------- mask -> count
__global__ void count_kernel(const unsigned char* __restrict__ mask,
                             int* __restrict__ counts) {
  __shared__ int red[256];
  int b = blockIdx.x;
  int t = threadIdx.x;
  int sum = 0;
  for (int s = t; s < S_; s += 256) sum += mask[b * S_ + s] ? 1 : 0;
  red[t] = sum; __syncthreads();
  for (int off2 = 128; off2 > 0; off2 >>= 1) {
    if (t < off2) red[t] += red[t + off2];
    __syncthreads();
  }
  if (t == 0) {
    int c = red[0];
    if (c < 1) c = 1;
    if (c > S_) c = S_;
    counts[b] = c;
  }
}

// ------------------------------------- gather chunks + cluster projections
__global__ void gather_proj(const unsigned short* __restrict__ kbf,
                            const unsigned short* __restrict__ vbf,
                            const int* __restrict__ table,
                            const int* __restrict__ counts,
                            const float* __restrict__ e_w,
                            const float* __restrict__ f_w,
                            unsigned short* __restrict__ kprojT,
                            unsigned short* __restrict__ vproj) {
  int idx = blockIdx.x * blockDim.x + threadIdx.x;   // (b,h,p,d) packed
  int d = idx & (DEPTH_ - 1);
  int p = (idx >> 6) & (P_ - 1);
  int h = (idx >> 13) & (H_ - 1);
  int b = idx >> 16;
  const int*   pos = table + (((long)(counts[b] - 1)) * P_ + p) * C_;
  const float* ew  = e_w + (h * P_ + p) * C_;
  const float* fw  = f_w + (h * P_ + p) * C_;
  float ks = 0.f, vs = 0.f;
  #pragma unroll
  for (int c = 0; c < C_; ++c) {
    int s = pos[c];
    if (s < S_) {   // index S == padded zero row
      long off = ((long)b * S_ + s) * DM_ + h * DEPTH_ + d;
      ks += bf2f(kbf[off]) * ew[c];
      vs += bf2f(vbf[off]) * fw[c];
    }
  }
  long bh = (long)b * H_ + h;
  kprojT[(bh * DEPTH_ + d) * P_ + p] = f2bf(ks);
  vproj [(bh * P_ + p) * DEPTH_ + d] = f2bf(vs);
}

// --------------------------- combined conv weight matrix (640 x 128) + bias
__global__ void build_mcat(const float* __restrict__ ck1, const float* __restrict__ cb1,
                           const float* __restrict__ ck3, const float* __restrict__ cb3,
                           const float* __restrict__ ck5, const float* __restrict__ cb5,
                           unsigned short* __restrict__ mcat,
                           float* __restrict__ biasScalar) {
  int i = blockIdx.x * blockDim.x + threadIdx.x;
  if (i == 0) biasScalar[0] = (cb1[0] + cb3[0] + cb5[0]) * (1.f / 3.f);
  if (i >= 5 * P_ * P_) return;
  int p  = i & (P_ - 1);
  int j  = (i >> 7) & (P_ - 1);
  int dh = i >> 14;
  int c  = j - p + 63;              // XLA SAME, cross-correlation
  float v = 0.f;
  if (c >= 0 && c < P_) {
    v = ck5[dh * P_ + c];
    if (dh >= 1 && dh <= 3) v += ck3[(dh - 1) * P_ + c];
    if (dh == 2)            v += ck1[c];
  }
  mcat[(dh * P_ + j) * P_ + p] = f2bf(v * (1.f / 3.f));
}

// ------------------------------------------------- softmax over P (in place)
__global__ void softmax128(unsigned short* __restrict__ data) {
  int lane = threadIdx.x & 31;
  int wid  = threadIdx.x >> 5;
  long row = (long)blockIdx.x * 8 + wid;
  unsigned short* ptr = data + row * P_;
  float v[4];
  #pragma unroll
  for (int i = 0; i < 4; ++i) v[i] = bf2f(ptr[lane * 4 + i]);
  float m = fmaxf(fmaxf(v[0], v[1]), fmaxf(v[2], v[3]));
  #pragma unroll
  for (int off2 = 16; off2 > 0; off2 >>= 1) m = fmaxf(m, __shfl_xor(m, off2, 32));
  float s = 0.f;
  #pragma unroll
  for (int i = 0; i < 4; ++i) { v[i] = __expf(v[i] - m); s += v[i]; }
  #pragma unroll
  for (int off2 = 16; off2 > 0; off2 >>= 1) s += __shfl_xor(s, off2, 32);
  float inv = 1.f / s;
  #pragma unroll
  for (int i = 0; i < 4; ++i) ptr[lane * 4 + i] = f2bf(v[i] * inv);
}

// ------------------------------------------------------------- WMMA GEMM
// C[z] = alpha * A[z] * B[z] (+ bias).  BM=128, BN=64, BK=32, 8 waves of 32x32.
// A tile staged by TDM (tensor_load_to_lds) unless convGather; B staged
// transposed by threads so B fragments are two ds_load_b128 per 16x16.
__global__ __launch_bounds__(256)
void gemm_bf16(const unsigned short* __restrict__ A,
               const unsigned short* __restrict__ Bm,
               float* __restrict__ C, unsigned short* __restrict__ Cbf,
               int M, int N, int K, int lda, int ldb, int ldc,
               int zInnerA, long aSO, long aSI,
               long bSO,
               int zInnerC, long cSO, long cSI,
               float alpha, const float* __restrict__ bias, int biasMode,
               int convGather) {
  __shared__ __align__(1024) unsigned short ldsA[128 * 32];   // row-major tile
  __shared__ __align__(1024) unsigned short ldsBT[64 * 32];   // transposed [n][k]

  const int t    = threadIdx.x;
  const int lane = t & 31, wid = t >> 5;
  const int wm   = wid & 3, wn = wid >> 2;    // 4x2 wave grid
  const int z    = blockIdx.z;
  const int rowBase = blockIdx.y * 128;
  const int colBase = blockIdx.x * 64;
  const long offA = (long)(z / zInnerA) * aSO + (long)(z % zInnerA) * aSI;
  const long offB = (long)z * bSO;
  const long offC = (long)(z / zInnerC) * cSO + (long)(z % zInnerC) * cSI;

  floatx8 acc[2][2] = {};

  const int arow = t >> 1, akc = (t & 1) * 16;   // manual A staging map
  const int bk = t & 31, bnb = (t >> 5) * 8;     // B staging map

  const unsigned ldsA_addr = (unsigned)(unsigned long long)(void*)&ldsA[0];

  for (int k0 = 0; k0 < K; k0 += 32) {
    // ---- stage A tile (128 x 32)
    if (!convGather) {
      if (wid == 0) {          // one wave drives the Tensor Data Mover
        tdm_load_2d(A + offA + (long)rowBase * lda + k0, ldsA_addr,
                    /*tileX*/ 32u, /*tileY*/ 128u, (unsigned long long)lda);
        __builtin_amdgcn_s_wait_tensorcnt(0);
      }
    } else {
      // virtual A[s][dh*128+j] = scores[z][s+dh-2][j], zero padded rows
      unsigned short tmp[16];
      int kk = k0 + akc;
      int dh = kk >> 7;
      int j0 = kk & 127;
      int srow = rowBase + arow + dh - 2;
      if (srow >= 0 && srow < S_) {
        const unsigned short* src = A + offA + (long)srow * P_ + j0;
        #pragma unroll
        for (int j = 0; j < 16; ++j) tmp[j] = src[j];
      } else {
        #pragma unroll
        for (int j = 0; j < 16; ++j) tmp[j] = 0;
      }
      #pragma unroll
      for (int j = 0; j < 16; ++j) ldsA[arow * 32 + akc + j] = tmp[j];
    }
    // ---- stage B tile (32 x 64), transposed into [n][k]
    #pragma unroll
    for (int j = 0; j < 8; ++j) {
      int n = bnb + j;
      ldsBT[n * 32 + bk] = Bm[offB + (long)(k0 + bk) * ldb + (colBase + n)];
    }
    if (k0 + 32 < K)   // prefetch next B tile rows (global_prefetch path)
      __builtin_prefetch(Bm + offB + (long)(k0 + 32 + bk) * ldb + colBase, 0, 1);
    __syncthreads();

    // ---- fragments per documented 16-bit 16x32 layout
    const int fm = lane & 15, hlf = lane >> 4;
    uintx8 au[2], bu[2];
    #pragma unroll
    for (int mi = 0; mi < 2; ++mi) {
      int r0 = wm * 32 + mi * 16 + fm;
      #pragma unroll
      for (int v = 0; v < 8; ++v) {
        int k2 = ((v < 4) ? v * 2 : 16 + (v - 4) * 2) + hlf * 8;
        au[mi][v] = *(const unsigned int*)&ldsA[r0 * 32 + k2];
      }
    }
    #pragma unroll
    for (int ni = 0; ni < 2; ++ni) {
      int c0 = wn * 32 + ni * 16 + fm;
      #pragma unroll
      for (int v = 0; v < 8; ++v) {
        int k2 = ((v < 4) ? v * 2 : 16 + (v - 4) * 2) + hlf * 8;
        bu[ni][v] = *(const unsigned int*)&ldsBT[c0 * 32 + k2];
      }
    }
    #pragma unroll
    for (int mi = 0; mi < 2; ++mi)
      #pragma unroll
      for (int ni = 0; ni < 2; ++ni) {
        bf16x16 af = __builtin_bit_cast(bf16x16, au[mi]);
        bf16x16 bf = __builtin_bit_cast(bf16x16, bu[ni]);
        acc[mi][ni] = __builtin_amdgcn_wmma_f32_16x16x32_bf16(
            false, af, false, bf, (short)0, acc[mi][ni], false, false);
      }
    __syncthreads();
  }

  // ---- epilogue: 8-VGPR f32 C/D layout (lane n, VGPR i -> row i + 8*half)
  const int fn = lane & 15, hlf = lane >> 4;
  #pragma unroll
  for (int mi = 0; mi < 2; ++mi)
    #pragma unroll
    for (int ni = 0; ni < 2; ++ni)
      #pragma unroll
      for (int i = 0; i < 8; ++i) {
        int row = rowBase + wm * 32 + mi * 16 + i + hlf * 8;
        int col = colBase + wn * 32 + ni * 16 + fn;
        float val = acc[mi][ni][i] * alpha;
        if (biasMode == 1)      val += bias[col];
        else if (biasMode == 2) val += bias[0];
        long o = offC + (long)row * ldc + col;
        if (C)   C[o]   = val;
        if (Cbf) Cbf[o] = f2bf(val);
      }
}

// ================================================================ launcher
extern "C" void kernel_launch(void* const* d_in, const int* in_sizes, int n_in,
                              void* d_out, int out_size, void* d_ws, size_t ws_size,
                              hipStream_t stream) {
  const float*         x     = (const float*)d_in[0];
  const unsigned char* mask  = (const unsigned char*)d_in[1];
  const float*         wq    = (const float*)d_in[2];
  const float*         wk    = (const float*)d_in[3];
  const float*         wv    = (const float*)d_in[4];
  const float*         e_w   = (const float*)d_in[5];
  const float*         f_w   = (const float*)d_in[6];
  const float*         ck1   = (const float*)d_in[7];
  const float*         cb1   = (const float*)d_in[8];
  const float*         ck3   = (const float*)d_in[9];
  const float*         cb3   = (const float*)d_in[10];
  const float*         ck5   = (const float*)d_in[11];
  const float*         cb5   = (const float*)d_in[12];
  const float*         dw    = (const float*)d_in[13];
  const float*         db    = (const float*)d_in[14];
  const int*           table = (const int*)d_in[15];
  (void)in_sizes; (void)n_in; (void)out_size; (void)ws_size;

  char* ws = (char*)d_ws;
  size_t off = 0;
  auto alloc = [&](size_t bytes) -> void* {
    void* p = ws + off;
    off = (off + bytes + 255) & ~(size_t)255;
    return p;
  };

  const size_t NX = (size_t)B_ * S_ * DM_;        // 16.78M
  const size_t NW = (size_t)DM_ * DM_;            // 262144

  unsigned short* x_bf     = (unsigned short*)alloc(NX * 2);
  unsigned short* wq_bf    = (unsigned short*)alloc(NW * 2);
  unsigned short* wk_bf    = (unsigned short*)alloc(NW * 2);
  unsigned short* wv_bf    = (unsigned short*)alloc(NW * 2);
  unsigned short* wd_bf    = (unsigned short*)alloc(NW * 2);
  unsigned short* q_bf     = (unsigned short*)alloc(NX * 2);  // reused as concat
  unsigned short* k_bf     = (unsigned short*)alloc(NX * 2);
  unsigned short* v_bf     = (unsigned short*)alloc(NX * 2);
  unsigned short* kprojT   = (unsigned short*)alloc((size_t)B_ * H_ * DEPTH_ * P_ * 2);
  unsigned short* vproj    = (unsigned short*)alloc((size_t)B_ * H_ * P_ * DEPTH_ * 2);
  unsigned short* mcat     = (unsigned short*)alloc((size_t)5 * P_ * P_ * 2);
  float*          biasScal = (float*)alloc(sizeof(float));
  unsigned short* scores   = (unsigned short*)alloc((size_t)B_ * H_ * S_ * P_ * 2);
  unsigned short* conv     = (unsigned short*)alloc((size_t)B_ * H_ * S_ * P_ * 2);
  int*            counts   = (int*)alloc(B_ * sizeof(int));
  unsigned short* concat   = q_bf;   // q consumed before attn epilogue writes

  // 1. fp32 -> bf16 conversions
  cvt_f32_bf16<<<(int)((NX + 255) / 256), 256, 0, stream>>>(x,  x_bf,  (int)NX);
  cvt_f32_bf16<<<(int)((NW + 255) / 256), 256, 0, stream>>>(wq, wq_bf, (int)NW);
  cvt_f32_bf16<<<(int)((NW + 255) / 256), 256, 0, stream>>>(wk, wk_bf, (int)NW);
  cvt_f32_bf16<<<(int)((NW + 255) / 256), 256, 0, stream>>>(wv, wv_bf, (int)NW);
  cvt_f32_bf16<<<(int)((NW + 255) / 256), 256, 0, stream>>>(dw, wd_bf, (int)NW);

  // 2. token counts per batch
  count_kernel<<<B_, 256, 0, stream>>>(mask, counts);

  // 3. Q/K/V projections: (32768 x 512) @ (512 x 512), bf16-out
  {
    dim3 g(DM_ / 64, (B_ * S_) / 128, 1);
    gemm_bf16<<<g, 256, 0, stream>>>(x_bf, wq_bf, nullptr, q_bf,
        B_ * S_, DM_, DM_, DM_, DM_, DM_,
        1, 0, 0,  0,  1, 0, 0,  1.0f, nullptr, 0, 0);
    gemm_bf16<<<g, 256, 0, stream>>>(x_bf, wk_bf, nullptr, k_bf,
        B_ * S_, DM_, DM_, DM_, DM_, DM_,
        1, 0, 0,  0,  1, 0, 0,  1.0f, nullptr, 0, 0);
    gemm_bf16<<<g, 256, 0, stream>>>(x_bf, wv_bf, nullptr, v_bf,
        B_ * S_, DM_, DM_, DM_, DM_, DM_,
        1, 0, 0,  0,  1, 0, 0,  1.0f, nullptr, 0, 0);
  }

  // 4. gather chunks + cluster projections (k_proj / v_proj)
  gather_proj<<<(B_ * H_ * P_ * DEPTH_) / 256, 256, 0, stream>>>(
      k_bf, v_bf, table, counts, e_w, f_w, kprojT, vproj);

  // 5. combined conv weight matrix + scalar bias
  build_mcat<<<(5 * P_ * P_ + 255) / 256, 256, 0, stream>>>(
      ck1, cb1, ck3, cb3, ck5, cb5, mcat, biasScal);

  // 6. scores = (q @ k_projT) / 8 per (b,h): M=2048 N=128 K=64, z=128
  {
    dim3 g(P_ / 64, S_ / 128, B_ * H_);
    gemm_bf16<<<g, 256, 0, stream>>>(q_bf, kprojT, nullptr, scores,
        S_, P_, DEPTH_, DM_, P_, P_,
        /*zInnerA*/ H_, (long)S_ * DM_, (long)DEPTH_,
        /*bSO*/ (long)DEPTH_ * P_,
        /*zInnerC*/ 1, (long)S_ * P_, 0,
        0.125f, nullptr, 0, 0);
  }

  // 7. conv branches as one shifted GEMM: M=2048 N=128 K=640, z=128
  {
    dim3 g(P_ / 64, S_ / 128, B_ * H_);
    gemm_bf16<<<g, 256, 0, stream>>>(scores, mcat, nullptr, conv,
        S_, P_, 5 * P_, P_, P_, P_,
        1, (long)S_ * P_, 0,
        /*bSO*/ 0,
        1, (long)S_ * P_, 0,
        1.0f, biasScal, 2, /*convGather*/ 1);
  }

  // 8. softmax over P, in place on conv
  softmax128<<<(B_ * H_ * S_) / 8, 256, 0, stream>>>(conv);

  // 9. attn = w @ v_proj per (b,h): M=2048 N=64 K=128 -> concat (b,s,h*64+d)
  {
    dim3 g(1, S_ / 128, B_ * H_);
    gemm_bf16<<<g, 256, 0, stream>>>(conv, vproj, nullptr, concat,
        S_, DEPTH_, P_, P_, DEPTH_, DM_,
        1, (long)S_ * P_, 0,
        /*bSO*/ (long)P_ * DEPTH_,
        /*zInnerC*/ H_, (long)S_ * DM_, (long)DEPTH_,
        1.0f, nullptr, 0, 0);
  }

  // 10. out = concat @ dw + db : fp32 result to d_out
  {
    dim3 g(DM_ / 64, (B_ * S_) / 128, 1);
    gemm_bf16<<<g, 256, 0, stream>>>(concat, wd_bf, (float*)d_out, nullptr,
        B_ * S_, DM_, DM_, DM_, DM_, DM_,
        1, 0, 0,  0,  1, 0, 0,  1.0f, db, 1, 0);
  }
}
